// SemanticMatchingModel_70841190580597
// MI455X (gfx1250) — matrix operand order repaired and checked by reading, not compiled, not run on hardware
//
#include <hip/hip_runtime.h>
#include <hip/hip_bf16.h>

// CDNA5 (gfx1250, wave32) implementation of the relation-grouped bilinear form:
//   energy_b = tL_b^T * M_{rel_b} * tR_b,   M_rel = sum_i rel_table[rel,i]*assoc[i]
// Pipeline (all on `stream`, graph-capture safe):
//   1) zero counts  2) histogram rels  3) 1-thread scan (elem/group offsets)
//   4) scatter permutation (bucket batch by relation)
//   5) build Mt[rel][k][j] in f16 (padded 304x320, k-major for contiguous B-frag loads)
//   6) energy kernel: 1 wave per 16 batch elements of one relation,
//      v_wmma_f32_16x16x32_f16 over the 300-dim contraction, fused tR dot.
//      B-fragments are software double-buffered so each WMMA overlaps the next
//      fragment's L2 fetch (wait loadcnt<=2 instead of 0); tR gather is branchless.
// Workspace (~9.8 MB): ints [counts 40 | cursors 40 | elem_off 41 | group_off 41 | perm B]
// then f16 Mt[40][304][320].

typedef __attribute__((ext_vector_type(16))) _Float16 v16h;
typedef __attribute__((ext_vector_type(8)))  float    v8f;

#define TERM_DIM 300
#define REL_DIM  5
#define N_RELS   40
#define JPAD     320   // contraction dim padded to 10 tiles of K=32
#define KPAD     304   // output-column dim padded to 19 tiles of N=16
#define JTILES   10
#define KTILES   19

__global__ void zero_counts_kernel(int* counts) {
  int t = threadIdx.x;
  if (t < N_RELS) counts[t] = 0;
}

__global__ void count_kernel(const int* __restrict__ rels, int* counts, int B) {
  int b = blockIdx.x * blockDim.x + threadIdx.x;
  if (b < B) atomicAdd(&counts[rels[b]], 1);
}

__global__ void scan_kernel(const int* __restrict__ counts, int* cursors,
                            int* elem_off, int* group_off) {
  if (threadIdx.x == 0 && blockIdx.x == 0) {
    int eo = 0, go = 0;
    elem_off[0] = 0; group_off[0] = 0;
    for (int r = 0; r < N_RELS; ++r) {
      cursors[r] = eo;
      eo += counts[r];
      go += (counts[r] + 15) >> 4;
      elem_off[r + 1]  = eo;
      group_off[r + 1] = go;
    }
  }
}

__global__ void scatter_kernel(const int* __restrict__ rels, int* cursors,
                               int* __restrict__ perm, int B) {
  int b = blockIdx.x * blockDim.x + threadIdx.x;
  if (b < B) {
    int pos = atomicAdd(&cursors[rels[b]], 1);
    perm[pos] = b;
  }
}

// Mt[rel][k][j] (f16, KPAD x JPAD, zero padded) = sum_i rel_table[rel,i]*assoc[i,j,k]
__global__ void build_M_kernel(const float* __restrict__ rel_table,
                               const float* __restrict__ assoc,
                               _Float16* __restrict__ Mt) {
  int tid = blockIdx.x * blockDim.x + threadIdx.x;
  const int total = N_RELS * KPAD * JPAD;
  if (tid >= total) return;
  int rel = tid / (KPAD * JPAD);
  int rem = tid % (KPAD * JPAD);
  int k = rem / JPAD, j = rem % JPAD;
  float v = 0.f;
  if (j < TERM_DIM && k < TERM_DIM) {
    #pragma unroll
    for (int i = 0; i < REL_DIM; ++i)
      v += rel_table[rel * REL_DIM + i] * assoc[(i * TERM_DIM + j) * TERM_DIM + k];
  }
  Mt[tid] = (_Float16)v;
}

// B-fragment load: 32x16 f16 B layout (ISA 7.12.2 / sparse-B analogy):
// lane(half,n): VGPR p holds K(=j) = (p>=4?16:0) + 8*half + 2*(p&3) + {0,1}.  For our k-major Mt
// this is two runs of 8 contiguous f16 per lane.
__device__ __forceinline__ v16h load_bfrag(const _Float16* __restrict__ Mrel,
                                           int kt, int jt, int half, int hsub) {
  const _Float16* p = Mrel + (long long)(kt * 16 + hsub) * JPAD + jt * 32;
  v16h b;
  #pragma unroll
  for (int q = 0; q < 16; ++q) b[q] = p[((q >> 3) << 4) + half * 8 + (q & 7)];
  return b;
}

// One wave32 per group of 16 batch elements sharing one relation.
__global__ __launch_bounds__(256) void energy_kernel(
    const int*   __restrict__ terms_L, const int* __restrict__ terms_R,
    const float* __restrict__ term_table,
    const _Float16* __restrict__ Mt,
    const int* __restrict__ perm, const int* __restrict__ elem_off,
    const int* __restrict__ group_off, float* __restrict__ out)
{
  const int lane = threadIdx.x & 31;
  const int wave = threadIdx.x >> 5;
  const int w = blockIdx.x * 8 + wave;            // global group id (wave-uniform)
  const int total_groups = group_off[N_RELS];
  if (w >= total_groups) return;                  // whole wave exits together

  // find this group's relation (wave-uniform scan over 40 offsets)
  int rel = 0;
  while (group_off[rel + 1] <= w) ++rel;
  const int lg     = w - group_off[rel];
  const int gstart = elem_off[rel] + lg * 16;
  const int cnt    = min(16, elem_off[rel + 1] - gstart);

  const int half = lane >> 4;   // 0: lanes 0-15, 1: lanes 16-31
  const int hsub = lane & 15;   // A: row m ; B/C: column n

  // A-matrix row (tL embedding) for this lane's m = hsub (clamped for partial groups)
  const int eA = perm[gstart + (hsub < cnt ? hsub : 0)];
  const long long rL = (long long)terms_L[eA] * TERM_DIM;

  // tR rows + output indices for the 8 C-rows this lane owns (M = v + 8*half)
  long long rR[8];
  int pm[8], mrow[8];
  #pragma unroll
  for (int v = 0; v < 8; ++v) {
    int m = v + 8 * half;
    mrow[v] = m;
    int e = perm[gstart + (m < cnt ? m : 0)];
    pm[v] = e;
    rR[v] = (long long)terms_R[e] * TERM_DIM;
  }

  // Hoist all 10 A fragments (ISA 16-bit A 16x32 layout):
  // lane(half,m): VGPR p holds K = (p>=4?16:0) + 8*half + 2*(p&3) + {0,1}
  v16h aF[JTILES];
  #pragma unroll
  for (int jt = 0; jt < JTILES; ++jt) {
    #pragma unroll
    for (int p = 0; p < 8; ++p) {
      int K0 = ((p >> 2) << 4) + (half << 3) + ((p & 3) << 1);
      int j = jt * 32 + K0;
      float a0 = (j     < TERM_DIM) ? term_table[rL + j]     : 0.f;
      float a1 = (j + 1 < TERM_DIM) ? term_table[rL + j + 1] : 0.f;
      aF[jt][2 * p]     = (_Float16)a0;
      aF[jt][2 * p + 1] = (_Float16)a1;
    }
  }

  float partial[8];
  #pragma unroll
  for (int v = 0; v < 8; ++v) partial[v] = 0.f;

  const _Float16* Mrel = Mt + (long long)rel * KPAD * JPAD;

  // Software-pipelined B fragments: loads for step s+1 are issued before the
  // WMMA of step s (including across ktile boundaries), so each WMMA only has
  // to wait for its own already-inflight loads.
  v16h bcur = load_bfrag(Mrel, 0, 0, half, hsub);

  for (int kt = 0; kt < KTILES; ++kt) {
    v8f c = {0.f, 0.f, 0.f, 0.f, 0.f, 0.f, 0.f, 0.f};
    #pragma unroll
    for (int jt = 0; jt < JTILES; ++jt) {
      v16h bnext = bcur;
      if (jt + 1 < JTILES) {
        bnext = load_bfrag(Mrel, kt, jt + 1, half, hsub);
      } else if (kt + 1 < KTILES) {               // wave-uniform scalar branch
        bnext = load_bfrag(Mrel, kt + 1, 0, half, hsub);
      }
      c = __builtin_amdgcn_wmma_f32_16x16x32_f16(false, aF[jt], false, bcur,
                                                 (short)0, c, false, false);
      bcur = bnext;
    }
    // C layout: VGPR v holds M = v + 8*half, N = hsub. Fuse the tR dot.
    // Branchless tail masking: clamp the gather address, zero the value.
    const int   k   = kt * 16 + hsub;
    const int   kc  = (k < TERM_DIM) ? k : (TERM_DIM - 1);
    const float msk = (k < TERM_DIM) ? 1.f : 0.f;
    #pragma unroll
    for (int v = 0; v < 8; ++v) {
      float tr = term_table[rR[v] + kc] * msk;
      partial[v] += c[v] * tr;
    }
  }

  // reduce over the 16 N-lanes within each half (xor masks stay inside a half)
  #pragma unroll
  for (int v = 0; v < 8; ++v) {
    float s = partial[v];
    s += __shfl_xor(s, 1, 32);
    s += __shfl_xor(s, 2, 32);
    s += __shfl_xor(s, 4, 32);
    s += __shfl_xor(s, 8, 32);
    partial[v] = s;
  }
  if (hsub == 0) {  // lanes 0 (m=0..7) and 16 (m=8..15)
    #pragma unroll
    for (int v = 0; v < 8; ++v)
      if (mrow[v] < cnt) out[pm[v]] = partial[v];
  }
}

extern "C" void kernel_launch(void* const* d_in, const int* in_sizes, int n_in,
                              void* d_out, int out_size, void* d_ws, size_t ws_size,
                              hipStream_t stream) {
  const int*   rels       = (const int*)d_in[0];
  const int*   terms_L    = (const int*)d_in[1];
  const int*   terms_R    = (const int*)d_in[2];
  const float* term_table = (const float*)d_in[3];
  const float* rel_table  = (const float*)d_in[4];
  const float* assoc      = (const float*)d_in[5];
  float* out = (float*)d_out;
  const int B = in_sizes[0];

  int* wsi       = (int*)d_ws;
  int* counts    = wsi;         // 40
  int* cursors   = wsi + 64;    // 40
  int* elem_off  = wsi + 128;   // 41
  int* group_off = wsi + 192;   // 41
  int* perm      = wsi + 256;   // B
  size_t mt_off = (((size_t)(256 + B)) * 4 + 255) & ~(size_t)255;
  _Float16* Mt = (_Float16*)((char*)d_ws + mt_off);  // 40*304*320 f16 ~ 7.8 MB

  const int nb = (B + 255) / 256;
  zero_counts_kernel<<<1, 64, 0, stream>>>(counts);
  count_kernel<<<nb, 256, 0, stream>>>(rels, counts, B);
  scan_kernel<<<1, 1, 0, stream>>>(counts, cursors, elem_off, group_off);
  scatter_kernel<<<nb, 256, 0, stream>>>(rels, cursors, perm, B);

  const int mtot = N_RELS * KPAD * JPAD;
  build_M_kernel<<<(mtot + 255) / 256, 256, 0, stream>>>(rel_table, assoc, Mt);

  const int max_groups = (B + 15) / 16 + N_RELS;   // upper bound; extras exit early
  const int eblocks = (max_groups + 7) / 8;        // 8 wave32 per 256-thread block
  energy_kernel<<<eblocks, 256, 0, stream>>>(terms_L, terms_R, term_table, Mt,
                                             perm, elem_off, group_off, out);
}
